// MyLSTM_712964571843
// MI455X (gfx1250) — compile-verified, matrix operand order checked by reading, not run on hardware
//
#include <hip/hip_runtime.h>

typedef __attribute__((ext_vector_type(16))) __bf16 v16bf;
typedef __attribute__((ext_vector_type(8)))  __bf16 v8bf;
typedef __attribute__((ext_vector_type(8)))  float  v8f;
typedef __attribute__((ext_vector_type(4)))  unsigned int u32x4;
typedef __attribute__((ext_vector_type(8)))  int          i32x8;
typedef __attribute__((ext_vector_type(4)))  int          i32x4;

#define BATCH   4096
#define INF     2048        // K = input(1024) + hidden(1024)
#define HID     1024        // N per gate
#define BM      256         // batch rows per block (8 waves x 2 subtiles of 16)
#define BN      16          // hidden units per block (1 N-tile, x4 gates)
#define BK      64          // K per TDM tile = 2 wmma K-steps

__device__ __forceinline__ float sigf(float x) {
    return 1.0f / (1.0f + __expf(-x));
}

// ---- prepass: xh = concat(input, hx) converted to bf16, [BATCH][INF] ----
__global__ __launch_bounds__(256)
void lstm_cvt_xh(const float* __restrict__ inp, const float* __restrict__ hx,
                 __bf16* __restrict__ xh) {
    size_t i = (size_t)blockIdx.x * 256 + threadIdx.x;   // over BATCH*INF
    if (i >= (size_t)BATCH * INF) return;
    int row = (int)(i >> 11);
    int col = (int)(i & (INF - 1));
    float v = (col < 1024) ? inp[(size_t)row * 1024 + col]
                           : hx[(size_t)row * 1024 + (col - 1024)];
    xh[i] = (__bf16)v;
}

// ---- prepass: W_bf16[4][HID][INF] from Wi,Wf,Wg,Wo (fp32 [HID][INF]) ----
__global__ __launch_bounds__(256)
void lstm_cvt_w(const float* __restrict__ Wi, const float* __restrict__ Wf,
                const float* __restrict__ Wg, const float* __restrict__ Wo,
                __bf16* __restrict__ Wb) {
    size_t i = (size_t)blockIdx.x * 256 + threadIdx.x;   // over 4*HID*INF
    if (i >= (size_t)4 * HID * INF) return;
    int g = (int)(i >> 21);                 // HID*INF = 2^21
    size_t r = i & ((1u << 21) - 1);
    const float* src = (g == 0) ? Wi : (g == 1) ? Wf : (g == 2) ? Wg : Wo;
    Wb[i] = (__bf16)src[r];
}

__device__ __forceinline__ v16bf load_frag(const __bf16* base, int kb) {
    v8bf lo = *(const v8bf*)(base + kb);
    v8bf hi = *(const v8bf*)(base + 16 + kb);
    return __builtin_shufflevector(lo, hi,
            0, 1, 2, 3, 4, 5, 6, 7, 8, 9, 10, 11, 12, 13, 14, 15);
}

// ---- TDM: build D# descriptor and issue TENSOR_LOAD_TO_LDS ----
// D# per CDNA5 ISA 8.3/8.4: group0 = {count/type/addr/lds}, group1 = dims/strides,
// group2 = dim2 extension (for the 3-D weight tile), group3 unused.
// data_size=1 (2-byte elements). Strides in element units.
// clang-23 lane: 6-arg builtin (g0, g1, g2, g3, extra i32x8, cpol).
__device__ __forceinline__ void tdm_load(unsigned lds_off, const void* gptr,
                                         unsigned tile_d0, unsigned tile_d1,
                                         unsigned tile_d2,
                                         unsigned tens_d0, unsigned tens_d1,
                                         unsigned long long d0_stride,
                                         unsigned long long d1_stride,
                                         unsigned tens_d2) {
    unsigned long long ga = (unsigned long long)(uintptr_t)gptr;
    u32x4 g0;
    g0[0] = 1u;                                          // count=1, user mode
    g0[1] = lds_off;                                     // lds_addr (bytes)
    g0[2] = (unsigned)(ga & 0xffffffffu);                // global_addr[31:0]
    g0[3] = (unsigned)((ga >> 32) & 0x01ffffffu)         // global_addr[56:32]
          | (2u << 30);                                  // type=2 ("image")
    i32x8 g1;
    g1[0] = (int)(1u << 16);                             // data_size=1 (2B), mask=0
    g1[1] = (int)((tens_d0 & 0xffffu) << 16);            // tensor_dim0[15:0]
    g1[2] = (int)(((tens_d0 >> 16) & 0xffffu)            // tensor_dim0[31:16]
          | ((tens_d1 & 0xffffu) << 16));                // tensor_dim1[15:0]
    g1[3] = (int)(((tens_d1 >> 16) & 0xffffu)            // tensor_dim1[31:16]
          | ((tile_d0 & 0xffffu) << 16));                // tile_dim0
    g1[4] = (int)((tile_d1 & 0xffffu)                    // tile_dim1
          | ((tile_d2 & 0xffffu) << 16));                // tile_dim2
    g1[5] = (int)(d0_stride & 0xffffffffu);              // dim0_stride[31:0]
    g1[6] = (int)(((d0_stride >> 32) & 0xffffu)          // dim0_stride[47:32]
          | ((d1_stride & 0xffffu) << 16));              // dim1_stride[15:0]
    g1[7] = (int)((d1_stride >> 16) & 0xffffffffu);      // dim1_stride[47:16]
    i32x4 g2 = {(int)tens_d2, 0, 0, 0};                  // tensor_dim2 (3-D tile)
    i32x4 g3 = {0, 0, 0, 0};
    i32x8 gx = {0, 0, 0, 0, 0, 0, 0, 0};                 // extra group (unused)
    __builtin_amdgcn_tensor_load_to_lds(g0, g1, g2, g3, gx, 0);
}

// ---- fused GEMM (bf16 wmma, f32 acc) + LSTM epilogue, TDM-fed LDS ----
__global__ __launch_bounds__(256)
void lstm_gemm_epilogue(const __bf16* __restrict__ xh,   // [BATCH][INF]
                        const __bf16* __restrict__ Wb,   // [4][HID][INF]
                        const float* __restrict__ bi, const float* __restrict__ bfc,
                        const float* __restrict__ bg, const float* __restrict__ bo,
                        const float* __restrict__ cx,    // [BATCH][HID]
                        float* __restrict__ out) {       // h then c, each BATCH*HID
    __shared__ __bf16 ldsA[2][BM * BK];          // 2 x 32 KB
    __shared__ __bf16 ldsW[2][4 * BN * BK];      // 2 x  8 KB

    const int tid  = threadIdx.x;
    const int wave = tid >> 5;                   // 0..7, owns M rows wave*32..+31
    const int lane = tid & 31;
    const int m0   = blockIdx.x * BM;            // 16 blocks in x
    const int n0   = blockIdx.y * BN;            // 64 blocks in y

    // wmma fragment addressing (ISA 16-bit A/B layout)
    const int lm = lane & 15;
    const int kb = (lane >> 4) * 8;

    const __bf16* aTileBase = xh + (size_t)m0 * INF;
    const __bf16* wTileBase = Wb + (size_t)n0 * INF;

    v8f acc[2][4];                               // [M subtile][gate]
    #pragma unroll
    for (int s = 0; s < 2; ++s)
        #pragma unroll
        for (int g = 0; g < 4; ++g)
            acc[s][g] = (v8f){0.f, 0.f, 0.f, 0.f, 0.f, 0.f, 0.f, 0.f};

    const int NT = INF / BK;                     // 32 TDM tiles

    // issue DMA for tile 0 into buffer 0 (wave0: A-tile, wave1: W-tile)
    {
        if (wave == 0) {
            tdm_load((unsigned)(uintptr_t)&ldsA[0][0], aTileBase,
                     /*tile*/ BK, BM, 0,
                     /*tensor*/ INF, BATCH, /*strides*/ INF, 0, /*d2*/ 0);
        } else if (wave == 1) {
            // 3-D: x=K(64), y=n(16), z=gate(4); z advances by HID*INF elements
            tdm_load((unsigned)(uintptr_t)&ldsW[0][0], wTileBase,
                     /*tile*/ BK, BN, 4,
                     /*tensor*/ INF, HID, /*strides*/ INF,
                     (unsigned long long)HID * INF, /*d2*/ 4);
        }
    }

    for (int kt = 0; kt < NT; ++kt) {
        const int p = kt & 1;

        __builtin_amdgcn_s_wait_tensorcnt(0);    // DMA for tile kt landed (issuing waves)
        __syncthreads();                         // everyone sees buf p; prior reads of p^1 done

        if (kt + 1 < NT) {
            const size_t koff = (size_t)(kt + 1) * BK;
            if (wave == 0) {
                tdm_load((unsigned)(uintptr_t)&ldsA[p ^ 1][0], aTileBase + koff,
                         BK, BM, 0, INF, BATCH, INF, 0, 0);
            } else if (wave == 1) {
                tdm_load((unsigned)(uintptr_t)&ldsW[p ^ 1][0], wTileBase + koff,
                         BK, BN, 4, INF, HID, INF,
                         (unsigned long long)HID * INF, 4);
            }
        }

        // compute on buf p: 2 K-steps x 8 independent WMMAs
        #pragma unroll
        for (int ks = 0; ks < 2; ++ks) {
            const __bf16* Abase = &ldsA[p][0] + ks * 32;
            const __bf16* Wbase = &ldsW[p][0] + ks * 32;
            v16bf a0  = load_frag(Abase + (wave * 32 +      lm) * BK, kb);
            v16bf a1  = load_frag(Abase + (wave * 32 + 16 + lm) * BK, kb);
            v16bf bf0 = load_frag(Wbase + (0 * BN + lm) * BK, kb);
            v16bf bf1 = load_frag(Wbase + (1 * BN + lm) * BK, kb);
            v16bf bf2 = load_frag(Wbase + (2 * BN + lm) * BK, kb);
            v16bf bf3 = load_frag(Wbase + (3 * BN + lm) * BK, kb);

            acc[0][0] = __builtin_amdgcn_wmma_f32_16x16x32_bf16(false, a0, false, bf0, (short)0, acc[0][0], false, false);
            acc[1][0] = __builtin_amdgcn_wmma_f32_16x16x32_bf16(false, a1, false, bf0, (short)0, acc[1][0], false, false);
            acc[0][1] = __builtin_amdgcn_wmma_f32_16x16x32_bf16(false, a0, false, bf1, (short)0, acc[0][1], false, false);
            acc[1][1] = __builtin_amdgcn_wmma_f32_16x16x32_bf16(false, a1, false, bf1, (short)0, acc[1][1], false, false);
            acc[0][2] = __builtin_amdgcn_wmma_f32_16x16x32_bf16(false, a0, false, bf2, (short)0, acc[0][2], false, false);
            acc[1][2] = __builtin_amdgcn_wmma_f32_16x16x32_bf16(false, a1, false, bf2, (short)0, acc[1][2], false, false);
            acc[0][3] = __builtin_amdgcn_wmma_f32_16x16x32_bf16(false, a0, false, bf3, (short)0, acc[0][3], false, false);
            acc[1][3] = __builtin_amdgcn_wmma_f32_16x16x32_bf16(false, a1, false, bf3, (short)0, acc[1][3], false, false);
        }
    }

    // ---- epilogue: C/D layout -> M = 8*(lane>>4) + vgpr_idx, N = lane&15 ----
    const int n   = n0 + (lane & 15);
    const float bvi = bi[n], bvf = bfc[n], bvg = bg[n], bvo = bo[n];
    #pragma unroll
    for (int s = 0; s < 2; ++s) {
        const int mrow0 = m0 + wave * 32 + s * 16 + (lane >> 4) * 8;
        #pragma unroll
        for (int e = 0; e < 8; ++e) {
            const size_t idx = (size_t)(mrow0 + e) * HID + n;
            float iv = sigf(acc[s][0][e] + bvi);
            float fv = sigf(acc[s][1][e] + bvf);
            float gv = tanhf(acc[s][2][e] + bvg);
            float ov = sigf(acc[s][3][e] + bvo);
            float cn = fv * cx[idx] + iv * gv;
            out[idx]                       = ov * tanhf(cn);  // h_new
            out[(size_t)BATCH * HID + idx] = cn;              // c_new
        }
    }
}

extern "C" void kernel_launch(void* const* d_in, const int* in_sizes, int n_in,
                              void* d_out, int out_size, void* d_ws, size_t ws_size,
                              hipStream_t stream) {
    (void)in_sizes; (void)n_in; (void)out_size; (void)ws_size;
    const float* inp = (const float*)d_in[0];
    const float* hx  = (const float*)d_in[1];
    const float* cx  = (const float*)d_in[2];
    const float* Wi  = (const float*)d_in[3];
    const float* bi  = (const float*)d_in[4];
    const float* Wf  = (const float*)d_in[5];
    const float* bf_ = (const float*)d_in[6];
    const float* Wg  = (const float*)d_in[7];
    const float* bg  = (const float*)d_in[8];
    const float* Wo  = (const float*)d_in[9];
    const float* bo  = (const float*)d_in[10];
    float* out = (float*)d_out;

    __bf16* xh_bf = (__bf16*)d_ws;                                    // 16.8 MB
    __bf16* W_bf  = (__bf16*)((char*)d_ws + (size_t)BATCH * INF * 2); // 16.8 MB

    const int nxh = (BATCH * INF) / 256;        // 32768 blocks
    const int nw  = (4 * HID * INF) / 256;      // 32768 blocks
    lstm_cvt_xh<<<nxh, 256, 0, stream>>>(inp, hx, xh_bf);
    lstm_cvt_w <<<nw,  256, 0, stream>>>(Wi, Wf, Wg, Wo, W_bf);

    dim3 grid(BATCH / BM, HID / BN);            // 16 x 64
    lstm_gemm_epilogue<<<grid, 256, 0, stream>>>(xh_bf, W_bf, bi, bf_, bg, bo, cx, out);
}
